// PersonaMemoryMamba_21500606283977
// MI455X (gfx1250) — matrix operand order repaired and checked by reading, not compiled, not run on hardware
//
#include <hip/hip_runtime.h>
#include <math.h>

// ---------------------------------------------------------------------------
// PersonaMemoryMamba forward on gfx1250 (MI455X), bf16 WMMA for all big GEMMs.
// Shapes: B=128, T=512, E=768, D=256, P=256, DT=64.  BT = 65536 rows.
// Output layout (f32, concatenated): trajectory[128,512,64] | hidden[128,512,256]
//                                    | persona_vec[128,256] | alpha[128,1]
// 64-row M-tiles: each loaded B fragment feeds 4 WMMAs (register reuse),
// cutting L2 weight traffic 4x vs a 16-row tile.
// ---------------------------------------------------------------------------

typedef __attribute__((ext_vector_type(16))) __bf16 v16bf;
typedef __attribute__((ext_vector_type(8)))  float  v8f;

#define BT      65536
#define TRAJ_OFF 0
#define HID_OFF  4194304
#define PV_OFF   (4194304 + 16777216)
#define AL_OFF   (PV_OFF + 32768)

// workspace byte offsets
#define WS_WUPK  0        // 768*256 bf16  = 393216 B
#define WS_WEPK  393216   // 256*256 bf16  = 131072 B
#define WS_WT1PK 524288   // 256*128 bf16  = 65536 B
#define WS_WT2PK 589824   // 128*64  bf16  = 16384 B
#define WS_H0    606208   // 128*256 f32   = 131072 B

__device__ __forceinline__ v8f bf16_wmma(v16bf a, v16bf b, v8f c) {
  // D = A(16x32 bf16) * B(32x16 bf16) + C(16x16 f32)
  return __builtin_amdgcn_wmma_f32_16x16x32_bf16(
      /*neg_a=*/false, a, /*neg_b=*/false, b,
      /*c_mod=*/(short)0, c, /*reuse_a=*/false, /*reuse_b=*/false);
}

// A-matrix (16x32, 16-bit) fragment slot within a 512-element k-step block:
// lane 0-15 hold row M=lane, elems 0..7 -> K=0..7, elems 8..15 -> K=16..23;
// lane 16-31 hold row M=lane-16, elems 0..7 -> K=8..15, elems 8..15 -> K=24..31.
__device__ __forceinline__ int a_slot(int r, int kl) {
  int lane = r + (((kl >> 3) & 1) << 4);
  int e    = (kl & 7) + ((kl >> 4) << 3);
  return lane * 16 + e;
}

// --------------------------- weight packing --------------------------------
// Pack f32 weight [K,N] row-major into bf16 WMMA B-fragment order so each lane
// reads its 16 bf16 (32 B) contiguously:
//   B 32x16: lanes 0-15 -> N=lane, elems 0..7 -> K=0..7, elems 8..15 -> K=16..23;
//            lanes 16-31 -> N=lane-16, elems 0..7 -> K=8..15, elems 8..15 -> K=24..31.
__global__ __launch_bounds__(256) void k_pack(const float* __restrict__ src,
                                              __bf16* __restrict__ dst,
                                              int K, int N) {
  int idx = blockIdx.x * 256 + threadIdx.x;
  if (idx >= K * N) return;
  int k = idx / N, n = idx - k * N;
  int ntiles = N >> 4;
  int kt = k >> 5, kl = k & 31, nt = n >> 4, nl = n & 15;
  int lane = nl + (((kl >> 3) & 1) << 4);
  int e    = (kl & 7) + ((kl >> 4) << 3);
  size_t off = (((size_t)(kt * ntiles + nt)) * 32 + lane) * 16 + e;
  dst[off] = (__bf16)src[idx];
}

// --------------------------- persona head ----------------------------------
__global__ __launch_bounds__(256) void k_persona(
    const float* __restrict__ pcls, const float* __restrict__ Wp,
    const float* __restrict__ bp, const float* __restrict__ Wh0,
    const float* __restrict__ bh0, const float* __restrict__ h0_scale,
    const float* __restrict__ Wa1, const float* __restrict__ ba1,
    const float* __restrict__ ga, const float* __restrict__ bla,
    const float* __restrict__ Wa2, const float* __restrict__ ba2,
    const float* __restrict__ Wa3, const float* __restrict__ ba3,
    float* __restrict__ out_pv, float* __restrict__ out_alpha,
    float* __restrict__ h0) {
  __shared__ float sp[768];
  __shared__ float spv[256];
  __shared__ float sa1[128];
  __shared__ float sa2[64];
  __shared__ float redS[128], redQ[128];
  const int b = blockIdx.x, tid = threadIdx.x;
  for (int i = tid; i < 768; i += 256) sp[i] = pcls[b * 768 + i];
  __syncthreads();
  // persona_vec = pcls @ Wp + bp   [256]
  float acc = bp[tid];
  for (int k = 0; k < 768; ++k) acc = fmaf(sp[k], Wp[k * 256 + tid], acc);
  spv[tid] = acc;
  out_pv[b * 256 + tid] = acc;
  __syncthreads();
  // h0 = tanh(pv @ Wh0 + bh0) * scale
  float hacc = bh0[tid];
  for (int k = 0; k < 256; ++k) hacc = fmaf(spv[k], Wh0[k * 256 + tid], hacc);
  h0[b * 256 + tid] = tanhf(hacc) * h0_scale[0];
  // a1 = pv @ Wa1 + ba1   [128]
  if (tid < 128) {
    float s = ba1[tid];
    for (int k = 0; k < 256; ++k) s = fmaf(spv[k], Wa1[k * 128 + tid], s);
    sa1[tid] = s;
    redS[tid] = s;
    redQ[tid] = s * s;
  }
  __syncthreads();
  for (int st = 64; st >= 1; st >>= 1) {
    if (tid < st) { redS[tid] += redS[tid + st]; redQ[tid] += redQ[tid + st]; }
    __syncthreads();
  }
  const float m  = redS[0] * (1.0f / 128.0f);
  const float vv = redQ[0] * (1.0f / 128.0f) - m * m;
  const float rs = rsqrtf(vv + 1e-5f);
  if (tid < 128) {
    float v = (sa1[tid] - m) * rs * ga[tid] + bla[tid];
    sa1[tid] = fmaxf(v, 0.0f);   // ReLU(LN)
  }
  __syncthreads();
  if (tid < 64) {
    float s = ba2[tid];
    for (int k = 0; k < 128; ++k) s = fmaf(sa1[k], Wa2[k * 64 + tid], s);
    sa2[tid] = fmaxf(s, 0.0f);
  }
  __syncthreads();
  if (tid == 0) {
    float s = ba3[0];
    for (int k = 0; k < 64; ++k) s = fmaf(sa2[k], Wa3[k], s);
    out_alpha[b] = 1.0f / (1.0f + expf(-s));
  }
}

// ------------------ fused utt@Wu -> @We -> LN -> tanh ----------------------
// One workgroup (8 waves) per 64-row tile.  Each wave owns two 16-wide N-tiles
// across four 16-row M-blocks -> every B fragment loaded once feeds 4 WMMAs.
// GEMM1: K=768 (24 k-steps, A staged in a rolling 2-k-step LDS chunk).
// GEMM2: K=256 (8 k-steps) from a bf16 fragment-ordered x tile in LDS.
// LayerNorm+tanh done per 16-row group through a reused f32 staging buffer.
__global__ __launch_bounds__(256) void k_emlp(
    const float* __restrict__ utt, const __bf16* __restrict__ WuPk,
    const float* __restrict__ bu, const __bf16* __restrict__ WePk,
    const float* __restrict__ be, const float* __restrict__ ge,
    const float* __restrict__ ble, float* __restrict__ hid) {
  __shared__ __align__(32) __bf16 As[2 * 4 * 512];   // 64 rows x 64 K chunk
  __shared__ __align__(32) __bf16 Xs[8 * 4 * 512];   // 64x256 x, frag order
  __shared__ float Ys[16][257];
  __shared__ float redS[16][16], redQ[16][16];
  __shared__ float s_mean[16], s_rstd[16];

  const int tid  = threadIdx.x;
  const int lane = tid & 31;
  const int wave = tid >> 5;
  const int row0 = blockIdx.x * 64;
  const int n0   = wave * 2;
  const int colb = lane & 15, rowadd = (lane >> 4) << 3;

  v8f acc[4][2];
  #pragma unroll
  for (int mb = 0; mb < 4; ++mb) {
    acc[mb][0] = (v8f)0.0f;
    acc[mb][1] = (v8f)0.0f;
  }

  const v16bf* Bp = (const v16bf*)WuPk;  // 16 N-tiles
  for (int kc = 0; kc < 12; ++kc) {      // 12 chunks of 64 K (2 k-steps each)
    __syncthreads();
    // stage 64 rows x 64 K of utt -> bf16 A fragments
    for (int idx = tid; idx < 64 * 64; idx += 256) {
      int r = idx >> 6, kl64 = idx & 63;
      float v = utt[(size_t)(row0 + r) * 768 + (kc * 64 + kl64)];
      int mb = r >> 4, rl = r & 15;
      As[((kl64 >> 5) * 4 + mb) * 512 + a_slot(rl, kl64 & 31)] = (__bf16)v;
    }
    __syncthreads();
    #pragma unroll
    for (int k2 = 0; k2 < 2; ++k2) {
      const int kt = kc * 2 + k2;
      v16bf b0 = Bp[((size_t)kt * 16 + n0) * 32 + lane];
      v16bf b1 = Bp[((size_t)kt * 16 + n0 + 1) * 32 + lane];
      if (kt < 23)
        __builtin_prefetch(&Bp[((size_t)(kt + 1) * 16 + n0) * 32 + lane], 0, 1);
      #pragma unroll
      for (int mb = 0; mb < 4; ++mb) {
        v16bf a = *(const v16bf*)&As[(k2 * 4 + mb) * 512 + lane * 16];
        acc[mb][0] = bf16_wmma(a, b0, acc[mb][0]);
        acc[mb][1] = bf16_wmma(a, b1, acc[mb][1]);
      }
    }
  }
  // x = acc + bu -> Xs (bf16, fragment order; cols become K of GEMM2)
  #pragma unroll
  for (int mb = 0; mb < 4; ++mb) {
    #pragma unroll
    for (int half = 0; half < 2; ++half) {
      const int n = n0 + half;
      #pragma unroll
      for (int r = 0; r < 8; ++r) {
        int rl  = r + rowadd;
        int col = n * 16 + colb;
        float v = acc[mb][half][r] + bu[col];
        Xs[((col >> 5) * 4 + mb) * 512 + a_slot(rl, col & 31)] = (__bf16)v;
      }
    }
  }
  __syncthreads();

  // GEMM2: y = x @ We   (K=256)
  v8f acc2[4][2];
  #pragma unroll
  for (int mb = 0; mb < 4; ++mb) {
    acc2[mb][0] = (v8f)0.0f;
    acc2[mb][1] = (v8f)0.0f;
  }
  const v16bf* Bp2 = (const v16bf*)WePk;  // 16 N-tiles
  #pragma unroll
  for (int kt = 0; kt < 8; ++kt) {
    v16bf b0 = Bp2[((size_t)kt * 16 + n0) * 32 + lane];
    v16bf b1 = Bp2[((size_t)kt * 16 + n0 + 1) * 32 + lane];
    #pragma unroll
    for (int mb = 0; mb < 4; ++mb) {
      v16bf a = *(const v16bf*)&Xs[(kt * 4 + mb) * 512 + lane * 16];
      acc2[mb][0] = bf16_wmma(a, b0, acc2[mb][0]);
      acc2[mb][1] = bf16_wmma(a, b1, acc2[mb][1]);
    }
  }

  // row-wise LayerNorm(256) + tanh, one 16-row M-block at a time
  for (int g = 0; g < 4; ++g) {
    __syncthreads();
    #pragma unroll
    for (int half = 0; half < 2; ++half) {
      const int n = n0 + half;
      #pragma unroll
      for (int r = 0; r < 8; ++r) {
        int rl  = r + rowadd;
        int col = n * 16 + colb;
        Ys[rl][col] = acc2[g][half][r] + be[col];
      }
    }
    __syncthreads();
    const int row = tid >> 4, seg = tid & 15;
    float s = 0.0f, q = 0.0f;
    for (int i = 0; i < 16; ++i) {
      float v = Ys[row][seg * 16 + i];
      s += v;
      q += v * v;
    }
    redS[row][seg] = s;
    redQ[row][seg] = q;
    __syncthreads();
    if (seg == 0) {
      float S = 0.0f, Q = 0.0f;
      for (int i = 0; i < 16; ++i) { S += redS[row][i]; Q += redQ[row][i]; }
      float m  = S * (1.0f / 256.0f);
      float vv = Q * (1.0f / 256.0f) - m * m;
      s_mean[row] = m;
      s_rstd[row] = rsqrtf(vv + 1e-5f);
    }
    __syncthreads();
    const float m = s_mean[row], rs = s_rstd[row];
    for (int i = 0; i < 16; ++i) {
      int c = seg * 16 + i;
      float v = (Ys[row][c] - m) * rs * ge[c] + ble[c];
      hid[(size_t)(row0 + g * 16 + row) * 256 + c] = tanhf(v);
    }
  }
}

// --------------------------- EMA scan (in place) ---------------------------
__global__ __launch_bounds__(256) void k_scan(const float* __restrict__ alpha,
                                              const float* __restrict__ h0,
                                              float* hid) {
  const int g = blockIdx.x * 256 + threadIdx.x;  // 0..32767
  const int b = g >> 8;
  const int d = g & 255;
  const float a  = alpha[b];
  const float om = 1.0f - a;
  float h = h0[g];
  float* base = hid + (size_t)b * 512 * 256 + d;
  #pragma unroll 4
  for (int t = 0; t < 512; ++t) {
    float e = base[(size_t)t * 256];
    h = fmaf(a, h, om * e);
    base[(size_t)t * 256] = h;
  }
}

// --------------------- trajectory head + masking ---------------------------
// 64-row tiles; GEMM1 (K=256, N=128): wave owns 1 N-tile x 4 M-blocks.
// GEMM2 (K=128, N=64): wave owns (wave&3) N-tile x 2 M-blocks.
__global__ __launch_bounds__(256) void k_traj(
    float* hid, const unsigned char* __restrict__ mask,
    const __bf16* __restrict__ Wt1Pk, const float* __restrict__ bt1,
    const __bf16* __restrict__ Wt2Pk, const float* __restrict__ bt2,
    float* __restrict__ traj) {
  __shared__ __align__(32) __bf16 Hs[8 * 4 * 512];  // 64x256 hidden
  __shared__ __align__(32) __bf16 Gs[4 * 4 * 512];  // 64x128 gelu out
  __shared__ float s_m[64];
  const int tid  = threadIdx.x;
  const int lane = tid & 31;
  const int wave = tid >> 5;
  const int row0 = blockIdx.x * 64;
  const int colb = lane & 15, rowadd = (lane >> 4) << 3;

  for (int idx = tid; idx < 64 * 256; idx += 256) {
    int r = idx >> 8, c = idx & 255;
    int gr = row0 + r;                 // flat (b,t): gr == b*512 + t
    size_t o = (size_t)gr * 256 + c;
    float h  = hid[o];
    float mv = mask[gr] ? 1.0f : 0.0f;
    if (c == 0) s_m[r] = mv;
    hid[o] = h * mv;                   // masked hidden output, in place
    int mb = r >> 4, rl = r & 15;
    Hs[((c >> 5) * 4 + mb) * 512 + a_slot(rl, c & 31)] = (__bf16)h;
  }
  __syncthreads();

  // GEMM1 + exact GELU
  {
    const v16bf* Bp = (const v16bf*)Wt1Pk;  // 8 N-tiles
    v8f acc[4];
    #pragma unroll
    for (int mb = 0; mb < 4; ++mb) acc[mb] = (v8f)0.0f;
    #pragma unroll
    for (int kt = 0; kt < 8; ++kt) {
      v16bf b = Bp[((size_t)kt * 8 + wave) * 32 + lane];
      #pragma unroll
      for (int mb = 0; mb < 4; ++mb) {
        v16bf a = *(const v16bf*)&Hs[(kt * 4 + mb) * 512 + lane * 16];
        acc[mb] = bf16_wmma(a, b, acc[mb]);
      }
    }
    #pragma unroll
    for (int mb = 0; mb < 4; ++mb) {
      #pragma unroll
      for (int r = 0; r < 8; ++r) {
        int rl  = r + rowadd;
        int col = wave * 16 + colb;
        float x = acc[mb][r] + bt1[col];
        float g = 0.5f * x * (1.0f + erff(x * 0.70710678118654752f));
        Gs[((col >> 5) * 4 + mb) * 512 + a_slot(rl, col & 31)] = (__bf16)g;
      }
    }
  }
  __syncthreads();

  // GEMM2: all 8 waves; ntile = wave&3, M-blocks {mb0, mb0+1}
  {
    const int nt  = wave & 3;
    const int mb0 = (wave >> 2) * 2;
    const v16bf* Bp2 = (const v16bf*)Wt2Pk;  // 4 N-tiles
    v8f acc[2];
    acc[0] = (v8f)0.0f;
    acc[1] = (v8f)0.0f;
    #pragma unroll
    for (int kt = 0; kt < 4; ++kt) {
      v16bf b = Bp2[((size_t)kt * 4 + nt) * 32 + lane];
      #pragma unroll
      for (int j = 0; j < 2; ++j) {
        v16bf a = *(const v16bf*)&Gs[(kt * 4 + mb0 + j) * 512 + lane * 16];
        acc[j] = bf16_wmma(a, b, acc[j]);
      }
    }
    #pragma unroll
    for (int j = 0; j < 2; ++j) {
      #pragma unroll
      for (int r = 0; r < 8; ++r) {
        int rl  = r + rowadd;
        int row = (mb0 + j) * 16 + rl;
        int col = nt * 16 + colb;
        traj[(size_t)(row0 + row) * 64 + col] = (acc[j][r] + bt2[col]) * s_m[row];
      }
    }
  }
}

// ---------------------------------------------------------------------------
extern "C" void kernel_launch(void* const* d_in, const int* in_sizes, int n_in,
                              void* d_out, int out_size, void* d_ws,
                              size_t ws_size, hipStream_t stream) {
  (void)in_sizes; (void)n_in; (void)out_size; (void)ws_size;
  const float* persona_cls = (const float*)d_in[0];
  const float* utt_cls     = (const float*)d_in[1];
  const unsigned char* turn_mask = (const unsigned char*)d_in[2];
  const float* Wp  = (const float*)d_in[3];
  const float* bp  = (const float*)d_in[4];
  const float* Wh0 = (const float*)d_in[5];
  const float* bh0 = (const float*)d_in[6];
  const float* h0s = (const float*)d_in[7];
  const float* Wa1 = (const float*)d_in[8];
  const float* ba1 = (const float*)d_in[9];
  const float* ga  = (const float*)d_in[10];
  const float* bla = (const float*)d_in[11];
  const float* Wa2 = (const float*)d_in[12];
  const float* ba2 = (const float*)d_in[13];
  const float* Wa3 = (const float*)d_in[14];
  const float* ba3 = (const float*)d_in[15];
  const float* Wu  = (const float*)d_in[16];
  const float* bu  = (const float*)d_in[17];
  const float* We  = (const float*)d_in[18];
  const float* be  = (const float*)d_in[19];
  const float* ge  = (const float*)d_in[20];
  const float* ble = (const float*)d_in[21];
  const float* Wt1 = (const float*)d_in[22];
  const float* bt1 = (const float*)d_in[23];
  const float* Wt2 = (const float*)d_in[24];
  const float* bt2 = (const float*)d_in[25];

  float* out    = (float*)d_out;
  float* traj   = out + TRAJ_OFF;
  float* hid    = out + HID_OFF;
  float* opv    = out + PV_OFF;
  float* oalpha = out + AL_OFF;

  char* ws = (char*)d_ws;
  __bf16* WuPk  = (__bf16*)(ws + WS_WUPK);
  __bf16* WePk  = (__bf16*)(ws + WS_WEPK);
  __bf16* Wt1Pk = (__bf16*)(ws + WS_WT1PK);
  __bf16* Wt2Pk = (__bf16*)(ws + WS_WT2PK);
  float*  h0    = (float*)(ws + WS_H0);

  k_pack<<<(768 * 256 + 255) / 256, 256, 0, stream>>>(Wu, WuPk, 768, 256);
  k_pack<<<(256 * 256 + 255) / 256, 256, 0, stream>>>(We, WePk, 256, 256);
  k_pack<<<(256 * 128 + 255) / 256, 256, 0, stream>>>(Wt1, Wt1Pk, 256, 128);
  k_pack<<<(128 * 64 + 255) / 256, 256, 0, stream>>>(Wt2, Wt2Pk, 128, 64);

  k_persona<<<128, 256, 0, stream>>>(persona_cls, Wp, bp, Wh0, bh0, h0s, Wa1,
                                     ba1, ga, bla, Wa2, ba2, Wa3, ba3, opv,
                                     oalpha, h0);

  k_emlp<<<BT / 64, 256, 0, stream>>>(utt_cls, WuPk, bu, WePk, be, ge, ble,
                                      hid);

  k_scan<<<128, 256, 0, stream>>>(oalpha, h0, hid);

  k_traj<<<BT / 64, 256, 0, stream>>>(hid, turn_mask, Wt1Pk, bt1, Wt2Pk, bt2,
                                      traj);
}
